// RotarySelfAttention_31258771980672
// MI455X (gfx1250) — matrix operand order repaired and checked by reading, not compiled, not run on hardware
//
#include <hip/hip_runtime.h>

#define BATCH  2
#define SEQ    2048
#define DMODEL 1024
#define NHEADS 16
#define HDIM   64

typedef __attribute__((ext_vector_type(16))) __bf16   v16bf;
typedef __attribute__((ext_vector_type(8)))  __bf16   v8bf;
typedef __attribute__((ext_vector_type(8)))  float    v8f;
typedef __attribute__((ext_vector_type(4)))  unsigned u32x4;
typedef __attribute__((ext_vector_type(8)))  unsigned u32x8;

static __device__ __forceinline__ v16bf load16(const __bf16* p) {
    return *(const v16bf*)p;
}
// A-fragment style load: two contiguous 8-element (16B) runs
static __device__ __forceinline__ v16bf load8x2(const __bf16* p0, const __bf16* p1) {
    v8bf lo = *(const v8bf*)p0;
    v8bf hi = *(const v8bf*)p1;
    return __builtin_shufflevector(lo, hi, 0,1,2,3,4,5,6,7,8,9,10,11,12,13,14,15);
}
static __device__ __forceinline__ v8f wmma_bf16(v16bf a, v16bf b, v8f c) {
    // (neg_a, A, neg_b, B, c_mod, C, reuse_a, reuse_b)
    return __builtin_amdgcn_wmma_f32_16x16x32_bf16(false, a, false, b, (short)0, c, false, false);
}

// ---------------------------------------------------------------------------
// Tensor Data Mover: 2D tile load (bf16 elements) into LDS.
// Builds D# group0 (count/lds/global/type) + group1 (dims/tile/stride) per
// CDNA5 ISA 8.3/8.4 and issues tensor_load_to_lds with 2 SGPR groups
// (VADDR2/VADDR3 = NULL is legal for tensors up to 2D). Tracked by TENSORcnt.
//   dim0/tile0: contiguous (innermost) dimension, in elements
//   dim1/tile1: outer dimension; stride0 = elements between outer rows
// ---------------------------------------------------------------------------
static __device__ __forceinline__ void tdm_load_2d_bf16(
        unsigned lds_byte_off, const void* gptr,
        unsigned dim0, unsigned dim1,
        unsigned tile0, unsigned tile1, unsigned long long stride0) {
    unsigned long long ga = (unsigned long long)(uintptr_t)gptr;
    u32x4 g0;
    g0[0] = 1u;                                   // count=1, user descriptor
    g0[1] = lds_byte_off;                         // lds_addr
    g0[2] = (unsigned)ga;                         // global_addr[31:0]
    g0[3] = (unsigned)((ga >> 32) & 0x01FFFFFFu)  // global_addr[56:32]
          | (2u << 30);                           // type = 2 ("image")
    u32x8 g1;
    g1[0] = (1u << 16);                           // data_size=1 (2 bytes), no mask
    g1[1] = (dim0 & 0xFFFFu) << 16;               // tensor_dim0[15:0]
    g1[2] = (dim0 >> 16) | ((dim1 & 0xFFFFu) << 16);
    g1[3] = (dim1 >> 16) | (tile0 << 16);         // tile_dim0
    g1[4] = tile1;                                // tile_dim1 (tile_dim2 = 0)
    g1[5] = (unsigned)(stride0 & 0xFFFFFFFFull);  // tensor_dim0_stride[31:0]
    g1[6] = (unsigned)(stride0 >> 32);            // stride0[47:32], stride1=0
    g1[7] = 0u;
    asm volatile("tensor_load_to_lds %0, %1" :: "s"(g0), "s"(g1) : "memory");
}

// ---------------------------------------------------------------------------
// f32 -> bf16 conversion
// ---------------------------------------------------------------------------
__global__ void cvt_f32_bf16(const float* __restrict__ in, __bf16* __restrict__ out, int n) {
    int i = blockIdx.x * blockDim.x + threadIdx.x;
    int stride = gridDim.x * blockDim.x;
    for (; i < n; i += stride) out[i] = (__bf16)in[i];
}

// ---------------------------------------------------------------------------
// Fused QKV projection + bias + RoPE (q,k) / transpose (v)
//   grid: (B*L/32, H, 3)   block: 32 (one wave computes a 32x64 tile:
//   2 A-fragments x 4 B-fragments -> 8 WMMAs per 32-deep K step)
//   q,k out: [B,H,L,D] bf16 (post-RoPE)    v out: [B,H,D,L] bf16 (transposed)
// ---------------------------------------------------------------------------
__global__ void qkv_rope_kernel(const __bf16* __restrict__ xb,
                                const __bf16* __restrict__ wqb,
                                const __bf16* __restrict__ wkb,
                                const __bf16* __restrict__ wvb,
                                const float*  __restrict__ bq,
                                const float*  __restrict__ bk,
                                const float*  __restrict__ bv,
                                const float*  __restrict__ cosp,
                                const float*  __restrict__ sinp,
                                __bf16* __restrict__ qr,
                                __bf16* __restrict__ kr,
                                __bf16* __restrict__ vt) {
    const int lane  = threadIdx.x;
    const int col16 = lane & 15;
    const int half  = lane >> 4;
    const int rowBase = blockIdx.x * 32;          // flattened B*L row
    const int h     = blockIdx.y;
    const int sel   = blockIdx.z;                 // 0=q 1=k 2=v
    const int eBase = h * HDIM;

    const __bf16* wb   = (sel == 0) ? wqb : ((sel == 1) ? wkb : wvb);
    const float*  bias = (sel == 0) ? bq  : ((sel == 1) ? bk  : bv);

    v8f acc[2][4] = {};
    const __bf16* xrow0 = xb + (size_t)(rowBase + col16) * DMODEL;
    const __bf16* xrow1 = xb + (size_t)(rowBase + 16 + col16) * DMODEL;
    const __bf16* wcolBase = wb + (size_t)(eBase + col16) * DMODEL;

    #pragma unroll 2
    for (int kb = 0; kb < DMODEL; kb += 32) {
        v16bf a0 = load8x2(xrow0 + kb + 16 * half, xrow0 + kb + 8 + 16 * half);
        v16bf a1 = load8x2(xrow1 + kb + 16 * half, xrow1 + kb + 8 + 16 * half);
        const __bf16* wcol = wcolBase + kb + 16 * half;
        #pragma unroll
        for (int t = 0; t < 4; ++t) {
            v16bf bfr = load16(wcol + t * 16 * DMODEL);
            acc[0][t] = wmma_bf16(a0, bfr, acc[0][t]);
            acc[1][t] = wmma_bf16(a1, bfr, acc[1][t]);
        }
    }

    float bvv[4];
    #pragma unroll
    for (int t = 0; t < 4; ++t) bvv[t] = bias[eBase + 16 * t + col16];

    const int bidx = rowBase / SEQ;
    const int l0   = rowBase % SEQ;
    const size_t bh = (size_t)bidx * NHEADS + h;

    if (sel == 2) {
        // V: store transposed [bh][d][l]
        #pragma unroll
        for (int g = 0; g < 2; ++g) {
            #pragma unroll
            for (int r = 0; r < 8; ++r) {
                int l = l0 + 16 * g + r + 8 * half;
                #pragma unroll
                for (int t = 0; t < 4; ++t)
                    vt[(bh * HDIM + 16 * t + col16) * SEQ + l] =
                        (__bf16)(acc[g][t][r] + bvv[t]);
            }
        }
    } else {
        __bf16* dst = (sel == 0) ? qr : kr;
        #pragma unroll
        for (int g = 0; g < 2; ++g) {
            #pragma unroll
            for (int r = 0; r < 8; ++r) {
                int l = l0 + 16 * g + r + 8 * half;
                const float* cr = cosp + (size_t)l * HDIM;
                const float* sr = sinp + (size_t)l * HDIM;
                float v0 = acc[g][0][r] + bvv[0], v1 = acc[g][1][r] + bvv[1];
                float v2 = acc[g][2][r] + bvv[2], v3 = acc[g][3][r] + bvv[3];
                int d0 = col16, d1 = 16 + col16, d2 = 32 + col16, d3 = 48 + col16;
                // rotate_half: out[d] = x[d]*cos[d] - x[d+32]*sin[d]  (d < 32)
                //              out[d] = x[d]*cos[d] + x[d-32]*sin[d]  (d >= 32)
                float o0 = v0 * cr[d0] - v2 * sr[d0];
                float o1 = v1 * cr[d1] - v3 * sr[d1];
                float o2 = v2 * cr[d2] + v0 * sr[d2];
                float o3 = v3 * cr[d3] + v1 * sr[d3];
                __bf16* q = dst + (bh * SEQ + l) * HDIM;
                q[d0] = (__bf16)o0; q[d1] = (__bf16)o1;
                q[d2] = (__bf16)o2; q[d3] = (__bf16)o3;
            }
        }
    }
}

// ---------------------------------------------------------------------------
// Flash attention: one wave handles a 16-query tile for one (b,h).
// K/V 32-key chunks are DMA'd into LDS by the Tensor Data Mover with a
// 2-deep double buffer; compute for chunk i overlaps TDM for chunk i+1.
//   grid: (L/16, B*H)  block: 32
//   ctx out: [B,L,DM] bf16
// ---------------------------------------------------------------------------
__global__ void attn_kernel(const __bf16* __restrict__ qr,
                            const __bf16* __restrict__ kr,
                            const __bf16* __restrict__ vt,
                            const unsigned char* __restrict__ mask,
                            __bf16* __restrict__ ctx) {
    __shared__ __bf16 kbuf[2][32 * HDIM];   // [key][d]   4KB per buffer
    __shared__ __bf16 vbuf[2][HDIM * 32];   // [d][key]   4KB per buffer
    __shared__ __bf16 pbuf[16 * 32];        // P staging (C-layout -> A-layout)

    const int lane  = threadIdx.x;
    const int col16 = lane & 15;
    const int half  = lane >> 4;
    const int qt    = blockIdx.x;
    const int bh    = blockIdx.y;
    const int bidx  = bh / NHEADS;
    const int h     = bh % NHEADS;

    // Q A-fragments (kept in registers for the whole key sweep)
    const __bf16* qrow = qr + ((size_t)bh * SEQ + qt * 16 + col16) * HDIM;
    v16bf qa0 = load8x2(qrow +      16 * half, qrow + 8 + 16 * half);  // d 0..31
    v16bf qa1 = load8x2(qrow + 32 + 16 * half, qrow + 40 + 16 * half);  // d 32..63

    float m[8], lsum[8];
    #pragma unroll
    for (int r = 0; r < 8; ++r) { m[r] = -1e30f; lsum[r] = 0.f; }
    v8f o0 = {}, o1 = {}, o2 = {}, o3 = {};

    const unsigned char* mrow = mask + (size_t)bidx * SEQ;
    const __bf16* kbase = kr + (size_t)bh * SEQ * HDIM;
    const __bf16* vbase = vt + (size_t)bh * HDIM * SEQ;

    const unsigned kOff0 = (unsigned)(uintptr_t)&kbuf[0][0];
    const unsigned kOff1 = (unsigned)(uintptr_t)&kbuf[1][0];
    const unsigned vOff0 = (unsigned)(uintptr_t)&vbuf[0][0];
    const unsigned vOff1 = (unsigned)(uintptr_t)&vbuf[1][0];

    // Prologue: DMA chunk 0 into buffer 0.
    //  K tile: 32 keys x 64 d, row stride HDIM (contiguous rows)
    //  V tile: 64 d rows x 32 keys, row stride SEQ
    tdm_load_2d_bf16(kOff0, kbase, HDIM, SEQ, HDIM, 32, HDIM);
    tdm_load_2d_bf16(vOff0, vbase, SEQ, HDIM, 32, HDIM, SEQ);

    for (int kc = 0; kc < SEQ; kc += 32) {
        const int buf = (kc >> 5) & 1;
        if (kc + 32 < SEQ) {
            const unsigned kOff = buf ? kOff0 : kOff1;
            const unsigned vOff = buf ? vOff0 : vOff1;
            tdm_load_2d_bf16(kOff, kbase + (size_t)(kc + 32) * HDIM,
                             HDIM, SEQ, HDIM, 32, HDIM);
            tdm_load_2d_bf16(vOff, vbase + (kc + 32),
                             SEQ, HDIM, 32, HDIM, SEQ);
            // 2 newly issued outstanding; TDM completes in order per wave,
            // so <=2 means the current chunk's pair has landed.
            __builtin_amdgcn_s_wait_tensorcnt(2);
        } else {
            __builtin_amdgcn_s_wait_tensorcnt(0);
        }
        asm volatile("" ::: "memory");   // keep LDS reads below the wait

        // ---- scores: two 16-key tiles, Q(16x64) . K^T, from LDS ----
        v8f s0 = {}, s1 = {};
        const __bf16* kl0 = &kbuf[buf][(size_t)col16 * HDIM];
        const __bf16* kl1 = &kbuf[buf][(size_t)(16 + col16) * HDIM];
        s0 = wmma_bf16(qa0, load16(kl0 + 16 * half),      s0);
        s0 = wmma_bf16(qa1, load16(kl0 + 32 + 16 * half), s0);
        s1 = wmma_bf16(qa0, load16(kl1 + 16 * half),      s1);
        s1 = wmma_bf16(qa1, load16(kl1 + 32 + 16 * half), s1);

        const bool mk0 = mrow[kc + col16]      != 0;
        const bool mk1 = mrow[kc + 16 + col16] != 0;

        // ---- online softmax over this 32-key chunk ----
        #pragma unroll
        for (int r = 0; r < 8; ++r) {
            float a0 = mk0 ? s0[r] * 0.125f : -1e30f;   // 1/sqrt(64)
            float a1 = mk1 ? s1[r] * 0.125f : -1e30f;
            float mx = fmaxf(a0, a1);
            mx = fmaxf(mx, __shfl_xor(mx, 1, 32));
            mx = fmaxf(mx, __shfl_xor(mx, 2, 32));
            mx = fmaxf(mx, __shfl_xor(mx, 4, 32));
            mx = fmaxf(mx, __shfl_xor(mx, 8, 32));
            float mn    = fmaxf(m[r], mx);
            float alpha = __expf(m[r] - mn);
            m[r] = mn;
            float p0 = __expf(a0 - mn);
            float p1 = __expf(a1 - mn);
            float ps = p0 + p1;
            ps += __shfl_xor(ps, 1, 32);
            ps += __shfl_xor(ps, 2, 32);
            ps += __shfl_xor(ps, 4, 32);
            ps += __shfl_xor(ps, 8, 32);
            lsum[r] = lsum[r] * alpha + ps;
            o0[r] *= alpha; o1[r] *= alpha; o2[r] *= alpha; o3[r] *= alpha;
            // stage P into LDS in row-major [16][32]
            pbuf[(r + 8 * half) * 32 + col16]      = (__bf16)p0;
            pbuf[(r + 8 * half) * 32 + 16 + col16] = (__bf16)p1;
        }
        __syncthreads();

        // ---- P(16x32) . V(32x64), V from LDS ----
        const __bf16* prow = pbuf + col16 * 32;
        v16bf pa = load8x2(prow + 16 * half, prow + 8 + 16 * half);
        const __bf16* vl = &vbuf[buf][(size_t)col16 * 32 + 16 * half];
        o0 = wmma_bf16(pa, load16(vl),            o0);
        o1 = wmma_bf16(pa, load16(vl + 16 * 32),  o1);
        o2 = wmma_bf16(pa, load16(vl + 32 * 32),  o2);
        o3 = wmma_bf16(pa, load16(vl + 48 * 32),  o3);
        __syncthreads();
    }

    #pragma unroll
    for (int r = 0; r < 8; ++r) {
        float inv = 1.0f / lsum[r];
        int l = qt * 16 + r + 8 * half;
        __bf16* crow = ctx + ((size_t)bidx * SEQ + l) * DMODEL + h * HDIM;
        crow[col16]      = (__bf16)(o0[r] * inv);
        crow[16 + col16] = (__bf16)(o1[r] * inv);
        crow[32 + col16] = (__bf16)(o2[r] * inv);
        crow[48 + col16] = (__bf16)(o3[r] * inv);
    }
}

// ---------------------------------------------------------------------------
// Output projection: out = ctx @ wo^T + bo   (fp32 out), 32x64 tile per wave
//   grid: (B*L/32, DM/64)  block: 32
// ---------------------------------------------------------------------------
__global__ void out_proj_kernel(const __bf16* __restrict__ ctx,
                                const __bf16* __restrict__ wob,
                                const float*  __restrict__ bo,
                                float* __restrict__ out) {
    const int lane  = threadIdx.x;
    const int col16 = lane & 15;
    const int half  = lane >> 4;
    const int rowBase = blockIdx.x * 32;
    const int eBase   = blockIdx.y * 64;

    v8f acc[2][4] = {};
    const __bf16* arow0 = ctx + (size_t)(rowBase + col16) * DMODEL;
    const __bf16* arow1 = ctx + (size_t)(rowBase + 16 + col16) * DMODEL;
    const __bf16* wcolBase = wob + (size_t)(eBase + col16) * DMODEL;

    #pragma unroll 2
    for (int kb = 0; kb < DMODEL; kb += 32) {
        v16bf a0 = load8x2(arow0 + kb + 16 * half, arow0 + kb + 8 + 16 * half);
        v16bf a1 = load8x2(arow1 + kb + 16 * half, arow1 + kb + 8 + 16 * half);
        const __bf16* wcol = wcolBase + kb + 16 * half;
        #pragma unroll
        for (int t = 0; t < 4; ++t) {
            v16bf bfr = load16(wcol + t * 16 * DMODEL);
            acc[0][t] = wmma_bf16(a0, bfr, acc[0][t]);
            acc[1][t] = wmma_bf16(a1, bfr, acc[1][t]);
        }
    }

    float bvv[4];
    #pragma unroll
    for (int t = 0; t < 4; ++t) bvv[t] = bo[eBase + 16 * t + col16];
    #pragma unroll
    for (int g = 0; g < 2; ++g) {
        #pragma unroll
        for (int r = 0; r < 8; ++r) {
            float* orow = out + (size_t)(rowBase + 16 * g + r + 8 * half) * DMODEL + eBase;
            #pragma unroll
            for (int t = 0; t < 4; ++t)
                orow[16 * t + col16] = acc[g][t][r] + bvv[t];
        }
    }
}

// ---------------------------------------------------------------------------
extern "C" void kernel_launch(void* const* d_in, const int* in_sizes, int n_in,
                              void* d_out, int out_size, void* d_ws, size_t ws_size,
                              hipStream_t stream) {
    // inputs: x, attention_mask, cos, sin, wq, bq, wk, bk, wv, bv, wo, bo
    const float* x    = (const float*)d_in[0];
    const unsigned char* amask = (const unsigned char*)d_in[1];
    const float* cosp = (const float*)d_in[2];
    const float* sinp = (const float*)d_in[3];
    const float* wq   = (const float*)d_in[4];
    const float* bq   = (const float*)d_in[5];
    const float* wk   = (const float*)d_in[6];
    const float* bk   = (const float*)d_in[7];
    const float* wv   = (const float*)d_in[8];
    const float* bv   = (const float*)d_in[9];
    const float* wo   = (const float*)d_in[10];
    const float* bo   = (const float*)d_in[11];
    float* out = (float*)d_out;

    constexpr size_t NX = (size_t)BATCH * SEQ * DMODEL;  // 4,194,304
    constexpr size_t NW = (size_t)DMODEL * DMODEL;       // 1,048,576

    __bf16* xb  = (__bf16*)d_ws;
    __bf16* wqb = xb  + NX;
    __bf16* wkb = wqb + NW;
    __bf16* wvb = wkb + NW;
    __bf16* wob = wvb + NW;
    __bf16* qr  = wob + NW;
    __bf16* kr  = qr  + NX;
    __bf16* vt  = kr  + NX;
    __bf16* ctx = vt  + NX;

    // 1) f32 -> bf16 conversions
    cvt_f32_bf16<<<2048, 256, 0, stream>>>(x,  xb,  (int)NX);
    cvt_f32_bf16<<<1024, 256, 0, stream>>>(wq, wqb, (int)NW);
    cvt_f32_bf16<<<1024, 256, 0, stream>>>(wk, wkb, (int)NW);
    cvt_f32_bf16<<<1024, 256, 0, stream>>>(wv, wvb, (int)NW);
    cvt_f32_bf16<<<1024, 256, 0, stream>>>(wo, wob, (int)NW);

    // 2) QKV + bias + RoPE (q,k) / transpose (v)
    dim3 gq(BATCH * SEQ / 32, NHEADS, 3);
    qkv_rope_kernel<<<gq, 32, 0, stream>>>(xb, wqb, wkb, wvb, bq, bk, bv,
                                           cosp, sinp, qr, kr, vt);

    // 3) flash attention (TDM double-buffered K/V)
    dim3 ga(SEQ / 16, BATCH * NHEADS);
    attn_kernel<<<ga, 32, 0, stream>>>(qr, kr, vt, amask, ctx);

    // 4) output projection
    dim3 go(BATCH * SEQ / 32, DMODEL / 64);
    out_proj_kernel<<<go, 32, 0, stream>>>(ctx, wob, bo, out);
}
